// TrajectoryGenerator_10943576670610
// MI455X (gfx1250) — matrix operand order repaired and checked by reading, not compiled, not run on hardware
//
#include <hip/hip_runtime.h>
#include <cstdint>
#include <cstddef>

typedef __attribute__((ext_vector_type(16))) __bf16 v16bf;
typedef __attribute__((ext_vector_type(8)))  float  v8f;
typedef __attribute__((ext_vector_type(4)))  int    v4i;

#define B_TOT 2048
#define S_SC  128
#define P_SC  16
#define E_DIM 64
#define H_DIM 128
#define BD_DIM 1024
#define P1_DIM 512
#define T_PRED 12

static __device__ __forceinline__ float sigmf(float x) { return 1.0f / (1.0f + expf(-x)); }

#if __has_builtin(__builtin_amdgcn_global_load_async_to_lds_b128) && \
    __has_builtin(__builtin_amdgcn_s_wait_asynccnt)
#define HAS_ASYNC_LDS 1
#define GLOBAL_V4I(p) ((__attribute__((address_space(1))) v4i*)(p))
#define LDS_V4I(p)    ((__attribute__((address_space(3))) v4i*)(p))
#else
#define HAS_ASYNC_LDS 0
#endif

// ---------------------------------------------------------------------------
// Generic bf16 GEMM: C[M,N] (f32) = A[M,K](bf16, lda) * Bw[N,K](bf16, ldb)^T + bias
// Block tile 128x128, K-step 32, double-buffered LDS with async global->LDS
// staging when available. 256 threads = 8 waves (4 along M x 2 along N),
// each wave computes a 32x64 subtile = 2x4 WMMA 16x16 accumulators.
// M,N multiples of 128; K multiple of 32 (guaranteed by caller). bias != null.
// ---------------------------------------------------------------------------
#define GBM 128
#define GBN 128
#define GBK 32
#define LDSP 40   // padded row (bf16 units) -> 80B rows, 16B aligned, conflict-free

__global__ __launch_bounds__(256) void gemm_bf16(
    const __bf16* __restrict__ A, int lda,
    const __bf16* __restrict__ Bw, int ldb,
    float* __restrict__ C, int N, int K,
    const float* __restrict__ bias)
{
  __shared__ __bf16 As[2][GBM][LDSP];
  __shared__ __bf16 Bs[2][GBN][LDSP];

  const int tid  = threadIdx.x;
  const int lane = tid & 31;
  const int wave = tid >> 5;
  const int wm   = wave & 3;   // wave M index (0..3)
  const int wn   = wave >> 2;  // wave N index (0..1)
  const int mBase = blockIdx.y * GBM;
  const int nBase = blockIdx.x * GBN;
  const int half = lane >> 4;  // lane half (0/1)
  const int l16  = lane & 15;

  // Per-thread staging coordinates: 512 16B chunks per tile, 2 per thread.
  const int row0 = tid >> 2;                 // chunk set 0: rows 0..63
  const int row1 = (tid + 256) >> 2;         // chunk set 1: rows 64..127
  const int kc   = (tid & 3) * 8;

  v8f zero = {};
  v8f acc[2][4];
#pragma unroll
  for (int i = 0; i < 2; ++i)
#pragma unroll
    for (int j = 0; j < 4; ++j) acc[i][j] = zero;

  auto stage = [&](int buf, int k0) {
#if HAS_ASYNC_LDS
    __builtin_amdgcn_global_load_async_to_lds_b128(
        GLOBAL_V4I(A + (size_t)(mBase + row0) * lda + k0 + kc),
        LDS_V4I(&As[buf][row0][kc]), 0, 0);
    __builtin_amdgcn_global_load_async_to_lds_b128(
        GLOBAL_V4I(A + (size_t)(mBase + row1) * lda + k0 + kc),
        LDS_V4I(&As[buf][row1][kc]), 0, 0);
    __builtin_amdgcn_global_load_async_to_lds_b128(
        GLOBAL_V4I(Bw + (size_t)(nBase + row0) * ldb + k0 + kc),
        LDS_V4I(&Bs[buf][row0][kc]), 0, 0);
    __builtin_amdgcn_global_load_async_to_lds_b128(
        GLOBAL_V4I(Bw + (size_t)(nBase + row1) * ldb + k0 + kc),
        LDS_V4I(&Bs[buf][row1][kc]), 0, 0);
#else
    *reinterpret_cast<uint4*>(&As[buf][row0][kc]) =
        *reinterpret_cast<const uint4*>(A + (size_t)(mBase + row0) * lda + k0 + kc);
    *reinterpret_cast<uint4*>(&As[buf][row1][kc]) =
        *reinterpret_cast<const uint4*>(A + (size_t)(mBase + row1) * lda + k0 + kc);
    *reinterpret_cast<uint4*>(&Bs[buf][row0][kc]) =
        *reinterpret_cast<const uint4*>(Bw + (size_t)(nBase + row0) * ldb + k0 + kc);
    *reinterpret_cast<uint4*>(&Bs[buf][row1][kc]) =
        *reinterpret_cast<const uint4*>(Bw + (size_t)(nBase + row1) * ldb + k0 + kc);
#endif
  };

  const int nk = K / GBK;
  stage(0, 0);
#if HAS_ASYNC_LDS
  __builtin_amdgcn_s_wait_asynccnt(0);
#endif
  __syncthreads();

  int cur = 0;
  for (int kt = 0; kt < nk; ++kt) {
    if (kt + 1 < nk) stage(cur ^ 1, (kt + 1) * GBK);   // prefetch next tile

    // Build fragments per ISA 7.12.2 layouts (16-bit A 16x32, B 32x16).
    v16bf afr[2];
#pragma unroll
    for (int tm = 0; tm < 2; ++tm) {
      int row = wm * 32 + tm * 16 + l16;
#pragma unroll
      for (int v = 0; v < 4; ++v) {
        int ka = half * 8 + 2 * v;          // VGPR 0..3: K = half*8 + 2v
        afr[tm][2 * v]     = As[cur][row][ka];
        afr[tm][2 * v + 1] = As[cur][row][ka + 1];
        int kb = 16 + half * 8 + 2 * v;     // VGPR 4..7: K = 16 + half*8 + 2v
        afr[tm][2 * (v + 4)]     = As[cur][row][kb];
        afr[tm][2 * (v + 4) + 1] = As[cur][row][kb + 1];
      }
    }
    v16bf bfr[4];
#pragma unroll
    for (int tn = 0; tn < 4; ++tn) {
      int row = wn * 64 + tn * 16 + l16;    // N index
#pragma unroll
      for (int v = 0; v < 8; ++v) {
        int k = half * 8 + 2 * v + (v >= 4 ? 8 : 0);  // VGPR 0..3: K = half*8 + 2v; VGPR 4..7: K = 16 + half*8 + 2(v-4)
        bfr[tn][2 * v]     = Bs[cur][row][k];
        bfr[tn][2 * v + 1] = Bs[cur][row][k + 1];
      }
    }

#pragma unroll
    for (int tm = 0; tm < 2; ++tm)
#pragma unroll
      for (int tn = 0; tn < 4; ++tn)
        acc[tm][tn] = __builtin_amdgcn_wmma_f32_16x16x32_bf16(
            false, afr[tm], false, bfr[tn], (short)0, acc[tm][tn], false, false);

#if HAS_ASYNC_LDS
    __builtin_amdgcn_s_wait_asynccnt(0);   // my prefetch writes landed in LDS
#endif
    __syncthreads();                       // everyone's writes visible
    cur ^= 1;
  }

  // D layout: lane(0-15)->N=lane, M=r ; lane(16-31)->N=lane-16, M=8+r
#pragma unroll
  for (int tm = 0; tm < 2; ++tm)
#pragma unroll
    for (int tn = 0; tn < 4; ++tn) {
      int col = nBase + wn * 64 + tn * 16 + l16;
      float bv = bias[col];
#pragma unroll
      for (int r = 0; r < 8; ++r) {
        int rowm = mBase + wm * 32 + tm * 16 + half * 8 + r;
        C[(size_t)rowm * N + col] = acc[tm][tn][r] + bv;
      }
    }
}

// ---------------------------------------------------------------------------
// LSTM cell + hidden2pos + position update + next spatial embedding.
// One block per pedestrian, 128 threads (one per hidden unit).
// ---------------------------------------------------------------------------
__global__ __launch_bounds__(128) void lstm_step(
    const float* __restrict__ Wih, const float* __restrict__ Whh,
    const float* __restrict__ bih, const float* __restrict__ bhh,
    const float* __restrict__ Wh2p, const float* __restrict__ bh2p,
    const float* __restrict__ Wse, const float* __restrict__ bse,
    const float* __restrict__ h, float* __restrict__ c,
    float* __restrict__ decin, float* __restrict__ pos,
    __bf16* __restrict__ dh, float* __restrict__ pred, int t)
{
  __shared__ float ex[E_DIM], hx[H_DIM], red0[H_DIM], red1[H_DIM], relsh[2];
  const int b = blockIdx.x, tid = threadIdx.x;
  if (tid < E_DIM) ex[tid] = decin[b * E_DIM + tid];
  hx[tid] = h[(size_t)b * H_DIM + tid];
  __syncthreads();

  float gate[4];
#pragma unroll
  for (int q = 0; q < 4; ++q) {           // gate order i,f,g,o
    int idx = q * H_DIM + tid;
    float acc = bih[idx] + bhh[idx];
    const float* wi = Wih + (size_t)idx * E_DIM;
    for (int e = 0; e < E_DIM; ++e) acc += wi[e] * ex[e];
    const float* wh = Whh + (size_t)idx * H_DIM;
    for (int j = 0; j < H_DIM; ++j) acc += wh[j] * hx[j];
    gate[q] = acc;
  }
  float cv = c[(size_t)b * H_DIM + tid];
  float c2 = sigmf(gate[1]) * cv + sigmf(gate[0]) * tanhf(gate[2]);
  float h2 = sigmf(gate[3]) * tanhf(c2);
  c[(size_t)b * H_DIM + tid] = c2;
  dh[(size_t)b * (H_DIM + BD_DIM) + tid] = (__bf16)h2;   // concat slot 0..127

  red0[tid] = Wh2p[tid] * h2;
  red1[tid] = Wh2p[H_DIM + tid] * h2;
  __syncthreads();
  for (int s = 64; s > 0; s >>= 1) {
    if (tid < s) { red0[tid] += red0[tid + s]; red1[tid] += red1[tid + s]; }
    __syncthreads();
  }
  if (tid == 0) {
    float r0 = red0[0] + bh2p[0], r1 = red1[0] + bh2p[1];
    relsh[0] = r0; relsh[1] = r1;
    pred[(size_t)t * B_TOT * 2 + b * 2 + 0] = r0;
    pred[(size_t)t * B_TOT * 2 + b * 2 + 1] = r1;
    pos[b * 2 + 0] += r0;                  // curr_pos = rel_pos + last_pos
    pos[b * 2 + 1] += r1;
  }
  __syncthreads();
  if (tid < E_DIM)
    decin[b * E_DIM + tid] = bse[tid] + Wse[tid * 2] * relsh[0] + Wse[tid * 2 + 1] * relsh[1];
}

// ---------------------------------------------------------------------------
// Pool-net weight preprocessing:
//  Wc[ch]  = W_p1[ch, :64] @ W_ps  (rank-2 collapse of embedding path)
//  beff    = b_p1 + W_p1[ch,:64] @ b_ps
//  Wp1h    = bf16(W_p1[ch, 64:192])
// ---------------------------------------------------------------------------
__global__ void prep_poolw(const float* __restrict__ Wp1, const float* __restrict__ bp1,
                           const float* __restrict__ Wps, const float* __restrict__ bps,
                           float* __restrict__ Wc, float* __restrict__ beff,
                           __bf16* __restrict__ Wp1h)
{
  int ch = blockIdx.x * blockDim.x + threadIdx.x;
  if (ch >= P1_DIM) return;
  const float* row = Wp1 + (size_t)ch * (E_DIM + H_DIM);
  float c0 = 0.f, c1 = 0.f, bb = bp1[ch];
  for (int e = 0; e < E_DIM; ++e) {
    c0 += row[e] * Wps[e * 2];
    c1 += row[e] * Wps[e * 2 + 1];
    bb += row[e] * bps[e];
  }
  Wc[ch * 2] = c0; Wc[ch * 2 + 1] = c1; beff[ch] = bb;
  for (int k = 0; k < H_DIM; ++k) Wp1h[(size_t)ch * H_DIM + k] = (__bf16)row[E_DIM + k];
}

// out1[s,i,j,ch] = hidG[s*16+j,ch] + rel·Wc[ch] + beff[ch]    (n = 32768*512 exact)
__global__ void out1_kernel(const float* __restrict__ hidG, const float* __restrict__ pos,
                            const float* __restrict__ Wc, const float* __restrict__ beff,
                            float* __restrict__ out1)
{
  int n = blockIdx.x * 256 + threadIdx.x;
  int ch = n & (P1_DIM - 1);
  int row = n >> 9;                 // flattened (s,i,j)
  int j = row & 15, i = (row >> 4) & 15, s = row >> 8;
  float r0 = pos[(s * 16 + j) * 2]     - pos[(s * 16 + i) * 2];
  float r1 = pos[(s * 16 + j) * 2 + 1] - pos[(s * 16 + i) * 2 + 1];
  out1[n] = hidG[(size_t)(s * 16 + j) * P1_DIM + ch] + r0 * Wc[ch * 2] + r1 * Wc[ch * 2 + 1] + beff[ch];
}

// Per-scene BN stats: mean/var over 256 (i,j) rows per (s, ch). Grid (C/256, S).
__global__ void scene_stats(const float* __restrict__ X, int C,
                            float* __restrict__ mean, float* __restrict__ rstd)
{
  int ch = blockIdx.x * 256 + threadIdx.x;
  int s  = blockIdx.y;
  const float* base = X + (size_t)s * 256 * C + ch;
  float sum = 0.f, sq = 0.f;
  for (int e = 0; e < 256; ++e) { float v = base[(size_t)e * C]; sum += v; sq += v * v; }
  float m = sum * (1.0f / 256.0f);
  float var = sq * (1.0f / 256.0f) - m * m;
  mean[(size_t)s * C + ch] = m;
  rstd[(size_t)s * C + ch] = rsqrtf(var + 1e-5f);
}

// x1 = bf16(relu(bn(out1)))   (n = 32768*512 exact)
__global__ void p1_apply(const float* __restrict__ out1, const float* __restrict__ mean,
                         const float* __restrict__ rstd, const float* __restrict__ g,
                         const float* __restrict__ be, __bf16* __restrict__ x1)
{
  int n = blockIdx.x * 256 + threadIdx.x;
  int ch = n & (P1_DIM - 1);
  int s = n >> 17;                  // n / (256*512)
  float y = (out1[n] - mean[(size_t)s * P1_DIM + ch]) * rstd[(size_t)s * P1_DIM + ch] * g[ch] + be[ch];
  x1[n] = (__bf16)fmaxf(y, 0.0f);
}

// relu(bn(out2)) then max over j; write into concat buffer dh[:,128:]. Grid (BD/256, B).
__global__ void p2_apply_max(const float* __restrict__ out2, const float* __restrict__ mean,
                             const float* __restrict__ rstd, const float* __restrict__ g,
                             const float* __restrict__ be, __bf16* __restrict__ dh)
{
  int ch = blockIdx.x * 256 + threadIdx.x;
  int b  = blockIdx.y;
  int s  = b >> 4;
  float m = mean[(size_t)s * BD_DIM + ch], r = rstd[(size_t)s * BD_DIM + ch];
  float gg = g[ch], bb = be[ch];
  float mx = -1e30f;
  for (int j = 0; j < P_SC; ++j) {
    float v = out2[((size_t)b * P_SC + j) * BD_DIM + ch];
    v = fmaxf((v - m) * r * gg + bb, 0.0f);
    mx = fmaxf(mx, v);
  }
  dh[(size_t)b * (H_DIM + BD_DIM) + H_DIM + ch] = (__bf16)mx;
}

// Batch BN stats, two stages over R=2048 rows. Grid (C/128, 16), block 128.
__global__ void bstat_part(const float* __restrict__ X, int C,
                           float* __restrict__ pS, float* __restrict__ pQ)
{
  int ch = blockIdx.x * 128 + threadIdx.x;
  int r0 = blockIdx.y * 128;
  float s = 0.f, q = 0.f;
  for (int r = 0; r < 128; ++r) { float v = X[(size_t)(r0 + r) * C + ch]; s += v; q += v * v; }
  pS[(size_t)blockIdx.y * C + ch] = s;
  pQ[(size_t)blockIdx.y * C + ch] = q;
}
__global__ void bstat_fin(const float* __restrict__ pS, const float* __restrict__ pQ, int C,
                          float* __restrict__ mean, float* __restrict__ rstd)
{
  int ch = blockIdx.x * 128 + threadIdx.x;
  float s = 0.f, q = 0.f;
  for (int p = 0; p < 16; ++p) { s += pS[(size_t)p * C + ch]; q += pQ[(size_t)p * C + ch]; }
  float m = s * (1.0f / 2048.0f);
  float var = q * (1.0f / 2048.0f) - m * m;
  mean[ch] = m; rstd[ch] = rsqrtf(var + 1e-5f);
}

__global__ void bapply_bf16(const float* __restrict__ X, const float* __restrict__ mean,
                            const float* __restrict__ rstd, const float* __restrict__ g,
                            const float* __restrict__ be, __bf16* __restrict__ Y, int C)
{
  int n = blockIdx.x * 256 + threadIdx.x;
  int ch = n & (C - 1);
  float y = (X[n] - mean[ch]) * rstd[ch] * g[ch] + be[ch];
  Y[n] = (__bf16)fmaxf(y, 0.0f);
}
__global__ void bapply_f32(const float* __restrict__ X, const float* __restrict__ mean,
                           const float* __restrict__ rstd, const float* __restrict__ g,
                           const float* __restrict__ be, float* __restrict__ Y, int C)
{
  int n = blockIdx.x * 256 + threadIdx.x;
  int ch = n & (C - 1);
  float y = (X[n] - mean[ch]) * rstd[ch] * g[ch] + be[ch];
  Y[n] = fmaxf(y, 0.0f);
}

__global__ void copy_f32(const float* __restrict__ s, float* __restrict__ d, int n)
{ int i = blockIdx.x * 256 + threadIdx.x; if (i < n) d[i] = s[i]; }

__global__ void fill_zero(float* __restrict__ d, int n)
{ int i = blockIdx.x * 256 + threadIdx.x; if (i < n) d[i] = 0.0f; }

__global__ void cvt_bf16(const float* __restrict__ s, __bf16* __restrict__ d, int n)
{ int i = blockIdx.x * 256 + threadIdx.x; if (i < n) d[i] = (__bf16)s[i]; }

__global__ void init_decin(const float* __restrict__ lpr, const float* __restrict__ Wse,
                           const float* __restrict__ bse, float* __restrict__ decin)
{
  int b = blockIdx.x, e = threadIdx.x;
  decin[b * E_DIM + e] = bse[e] + Wse[e * 2] * lpr[b * 2] + Wse[e * 2 + 1] * lpr[b * 2 + 1];
}

// ---------------------------------------------------------------------------
extern "C" void kernel_launch(void* const* d_in, const int* in_sizes, int n_in,
                              void* d_out, int out_size, void* d_ws, size_t ws_size,
                              hipStream_t stream)
{
  (void)in_sizes; (void)n_in; (void)out_size; (void)ws_size;
  const float* last_pos     = (const float*)d_in[0];
  const float* last_pos_rel = (const float*)d_in[1];
  const float* h0   = (const float*)d_in[2];
  const float* c0   = (const float*)d_in[3];
  /* d_in[4] seq_start_end: uniform groups of 16, derived analytically */
  const float* W_se = (const float*)d_in[5];  const float* b_se = (const float*)d_in[6];
  const float* W_ih = (const float*)d_in[7];  const float* W_hh = (const float*)d_in[8];
  const float* b_ih = (const float*)d_in[9];  const float* b_hh = (const float*)d_in[10];
  const float* W_h2p= (const float*)d_in[11]; const float* b_h2p= (const float*)d_in[12];
  const float* W_ps = (const float*)d_in[13]; const float* b_ps = (const float*)d_in[14];
  const float* W_p1 = (const float*)d_in[15]; const float* b_p1 = (const float*)d_in[16];
  const float* g_p1 = (const float*)d_in[17]; const float* be_p1= (const float*)d_in[18];
  const float* W_p2 = (const float*)d_in[19]; const float* b_p2 = (const float*)d_in[20];
  const float* g_p2 = (const float*)d_in[21]; const float* be_p2= (const float*)d_in[22];
  const float* W_m1 = (const float*)d_in[23]; const float* b_m1 = (const float*)d_in[24];
  const float* g_m1 = (const float*)d_in[25]; const float* be_m1= (const float*)d_in[26];
  const float* W_m2 = (const float*)d_in[27]; const float* b_m2 = (const float*)d_in[28];
  const float* g_m2 = (const float*)d_in[29]; const float* be_m2= (const float*)d_in[30];

  float* out = (float*)d_out;

  char* ws = (char*)d_ws;
  size_t off = 0;
  auto alloc = [&](size_t bytes) -> void* {
    void* p = ws + off; off += (bytes + 255) & ~(size_t)255; return p;
  };
  float*  pos   = (float*) alloc((size_t)B_TOT * 2 * 4);
  float*  cst   = (float*) alloc((size_t)B_TOT * H_DIM * 4);
  float*  hst   = (float*) alloc((size_t)B_TOT * H_DIM * 4);
  float*  dec   = (float*) alloc((size_t)B_TOT * E_DIM * 4);
  __bf16* dh    = (__bf16*)alloc((size_t)B_TOT * (H_DIM + BD_DIM) * 2);
  float*  hidG  = (float*) alloc((size_t)B_TOT * P1_DIM * 4);
  float*  out1  = (float*) alloc((size_t)B_TOT * P_SC * P1_DIM * 4);   // 32768x512
  float*  mean1 = (float*) alloc((size_t)S_SC * P1_DIM * 4);
  float*  rstd1 = (float*) alloc((size_t)S_SC * P1_DIM * 4);
  __bf16* x1    = (__bf16*)alloc((size_t)B_TOT * P_SC * P1_DIM * 2);
  float*  out2  = (float*) alloc((size_t)B_TOT * P_SC * BD_DIM * 4);   // 32768x1024
  float*  mean2 = (float*) alloc((size_t)S_SC * BD_DIM * 4);
  float*  rstd2 = (float*) alloc((size_t)S_SC * BD_DIM * 4);
  float*  g1    = (float*) alloc((size_t)B_TOT * BD_DIM * 4);
  float*  partS = (float*) alloc((size_t)16 * BD_DIM * 4);
  float*  partQ = (float*) alloc((size_t)16 * BD_DIM * 4);
  float*  meanB = (float*) alloc((size_t)BD_DIM * 4);
  float*  rstdB = (float*) alloc((size_t)BD_DIM * 4);
  __bf16* a1    = (__bf16*)alloc((size_t)B_TOT * BD_DIM * 2);
  float*  g2    = (float*) alloc((size_t)B_TOT * H_DIM * 4);
  __bf16* Wp1hB = (__bf16*)alloc((size_t)P1_DIM * H_DIM * 2);
  __bf16* Wp2B  = (__bf16*)alloc((size_t)BD_DIM * P1_DIM * 2);
  __bf16* Wm1B  = (__bf16*)alloc((size_t)BD_DIM * (H_DIM + BD_DIM) * 2);
  __bf16* Wm2B  = (__bf16*)alloc((size_t)H_DIM * BD_DIM * 2);
  float*  Wc    = (float*) alloc((size_t)P1_DIM * 2 * 4);
  float*  beff  = (float*) alloc((size_t)P1_DIM * 4);
  float*  zbias = (float*) alloc((size_t)P1_DIM * 4);

  // ---- one-time prep (runs every launch; cheap, deterministic) ----
  copy_f32<<<(B_TOT * H_DIM + 255) / 256, 256, 0, stream>>>(h0, hst, B_TOT * H_DIM);
  copy_f32<<<(B_TOT * H_DIM + 255) / 256, 256, 0, stream>>>(c0, cst, B_TOT * H_DIM);
  copy_f32<<<(B_TOT * 2 + 255) / 256, 256, 0, stream>>>(last_pos, pos, B_TOT * 2);
  init_decin<<<B_TOT, E_DIM, 0, stream>>>(last_pos_rel, W_se, b_se, dec);
  cvt_bf16<<<(BD_DIM * P1_DIM + 255) / 256, 256, 0, stream>>>(W_p2, Wp2B, BD_DIM * P1_DIM);
  cvt_bf16<<<(BD_DIM * (H_DIM + BD_DIM) + 255) / 256, 256, 0, stream>>>(W_m1, Wm1B, BD_DIM * (H_DIM + BD_DIM));
  cvt_bf16<<<(H_DIM * BD_DIM + 255) / 256, 256, 0, stream>>>(W_m2, Wm2B, H_DIM * BD_DIM);
  prep_poolw<<<2, 256, 0, stream>>>(W_p1, b_p1, W_ps, b_ps, Wc, beff, Wp1hB);
  fill_zero<<<(P1_DIM + 255) / 256, 256, 0, stream>>>(zbias, P1_DIM);

  const int ROW1 = B_TOT * P_SC;  // 32768 pooled rows

  for (int t = 0; t < T_PRED; ++t) {
    // 1) LSTM cell + rel_pos + curr_pos + next spatial embedding; h2(bf16) -> dh[:, :128]
    lstm_step<<<B_TOT, H_DIM, 0, stream>>>(W_ih, W_hh, b_ih, b_hh, W_h2p, b_h2p,
                                           W_se, b_se, hst, cst, dec, pos, dh, out, t);
    // 2) hidG = h2 @ W_p1[:,64:]^T     [2048x128]x[128x512]
    gemm_bf16<<<dim3(P1_DIM / GBN, B_TOT / GBM), 256, 0, stream>>>(
        dh, H_DIM + BD_DIM, Wp1hB, H_DIM, hidG, P1_DIM, H_DIM, zbias);
    // 3) rank-2 expansion of pool layer-1 pre-activation
    out1_kernel<<<(ROW1 * P1_DIM) / 256, 256, 0, stream>>>(hidG, pos, Wc, beff, out1);
    // 4) per-scene BN + ReLU -> bf16
    scene_stats<<<dim3(P1_DIM / 256, S_SC), 256, 0, stream>>>(out1, P1_DIM, mean1, rstd1);
    p1_apply<<<(ROW1 * P1_DIM) / 256, 256, 0, stream>>>(out1, mean1, rstd1, g_p1, be_p1, x1);
    // 5) pool layer-2 GEMM  [32768x512]x[512x1024]  (dominant FLOPs)
    gemm_bf16<<<dim3(BD_DIM / GBN, ROW1 / GBM), 256, 0, stream>>>(
        x1, P1_DIM, Wp2B, P1_DIM, out2, BD_DIM, P1_DIM, b_p2);
    // 6) per-scene BN + ReLU + max over j -> concat buffer dh[:, 128:]
    scene_stats<<<dim3(BD_DIM / 256, S_SC), 256, 0, stream>>>(out2, BD_DIM, mean2, rstd2);
    p2_apply_max<<<dim3(BD_DIM / 256, B_TOT), 256, 0, stream>>>(out2, mean2, rstd2, g_p2, be_p2, dh);
    // 7) decoder MLP layer 1  [2048x1152]x[1152x1024]
    gemm_bf16<<<dim3(BD_DIM / GBN, B_TOT / GBM), 256, 0, stream>>>(
        dh, H_DIM + BD_DIM, Wm1B, H_DIM + BD_DIM, g1, BD_DIM, H_DIM + BD_DIM, b_m1);
    bstat_part<<<dim3(BD_DIM / 128, 16), 128, 0, stream>>>(g1, BD_DIM, partS, partQ);
    bstat_fin<<<BD_DIM / 128, 128, 0, stream>>>(partS, partQ, BD_DIM, meanB, rstdB);
    bapply_bf16<<<(B_TOT * BD_DIM) / 256, 256, 0, stream>>>(g1, meanB, rstdB, g_m1, be_m1, a1, BD_DIM);
    // 8) decoder MLP layer 2  [2048x1024]x[1024x128] -> new hidden state
    gemm_bf16<<<dim3(H_DIM / GBN, B_TOT / GBM), 256, 0, stream>>>(
        a1, BD_DIM, Wm2B, BD_DIM, g2, H_DIM, BD_DIM, b_m2);
    bstat_part<<<dim3(H_DIM / 128, 16), 128, 0, stream>>>(g2, H_DIM, partS, partQ);
    bstat_fin<<<H_DIM / 128, 128, 0, stream>>>(partS, partQ, H_DIM, meanB, rstdB);
    bapply_f32<<<(B_TOT * H_DIM) / 256, 256, 0, stream>>>(g2, meanB, rstdB, g_m2, be_m2, hst, H_DIM);
  }

  // hT tail of output: [1, B, H] after pred [T,B,2]
  copy_f32<<<(B_TOT * H_DIM + 255) / 256, 256, 0, stream>>>(
      hst, out + (size_t)T_PRED * B_TOT * 2, B_TOT * H_DIM);
}
